// ITAttention_59760174956723
// MI455X (gfx1250) — matrix operand order repaired and checked by reading, not compiled, not run on hardware
//
#include <hip/hip_runtime.h>
#include <hip/hip_bf16.h>

typedef __attribute__((ext_vector_type(16))) __bf16 v16bf;
typedef __attribute__((ext_vector_type(8)))  __bf16 v8bf;
typedef __attribute__((ext_vector_type(4)))  __bf16 v4bf;
typedef __attribute__((ext_vector_type(8)))  float  v8f;

#define BT   128   // block tile (M and N)
#define KT   64    // K depth per stage
#define LDSS 72    // padded LDS row stride in bf16 (144B, 16B aligned)

// ---- CDNA5 async global->LDS copy (ASYNCcnt tracked) ----------------------
__device__ __forceinline__ void async_cp16(unsigned lds_off, const __bf16* gp) {
    asm volatile("global_load_async_to_lds_b128 %0, %1, off"
                 :: "v"(lds_off), "v"((unsigned long long)(size_t)gp)
                 : "memory");
}
__device__ __forceinline__ void wait_async0() {
    asm volatile("s_wait_asynccnt 0x0" ::: "memory");
}

// Stage a BT x KT bf16 tile into LDS.
//  KC=true : k-contiguous in memory -> async 16B chunks (4 per thread).
//  KC=false: k-strided but row-contiguous (sRow==1) -> vector load + transpose scatter.
template<bool KC>
__device__ __forceinline__ void stage_tile(const __bf16* __restrict__ G,
                                           long sRow, long sK,
                                           __bf16 (*S)[LDSS],
                                           long r0, long k0, int t)
{
    if (KC) {
        #pragma unroll
        for (int i = 0; i < 4; ++i) {
            const int ci = i * 256 + t;          // 1024 chunks of 16B
            const int r  = ci >> 3;              // 0..127
            const int c  = (ci & 7) * 8;         // 0..56
            async_cp16((unsigned)(unsigned long long)&S[r][c],
                       G + (r0 + r) * sRow + (k0 + c));
        }
    } else {
        #pragma unroll
        for (int i = 0; i < 4; ++i) {
            const int ci = i * 256 + t;          // 64 cols x 16 row-groups
            const int c  = ci >> 4;              // 0..63
            const int r  = (ci & 15) * 8;        // 0..120
            const v8bf v = *(const v8bf*)(G + (r0 + r) + (k0 + c) * sK);
            #pragma unroll
            for (int j = 0; j < 8; ++j) S[r + j][c] = v[j];
        }
    }
}

// ---------------------------------------------------------------------------
// Batched strided GEMM, bf16 WMMA, f32 accumulate, fused bias, bf16/f32 out.
//   C[z,m,n] = sum_k A[z*batchA + m*sAm + k*sAk] * B[z*batchB + k*sBk + n*sBn]
// Grid: (N/128, M/128, nbatch). Block: 256 threads = 8 waves.
// Wave layout: 2 rows x 4 cols of waves; each wave owns 64(m) x 32(n)
// = 4x2 WMMA 16x16 accumulators.
// ---------------------------------------------------------------------------
template<bool AKC, bool BKC, bool OUTBF>
__global__ __launch_bounds__(256)
void ita_gemm(const __bf16* __restrict__ A, const __bf16* __restrict__ Bm,
              const float* __restrict__ bias, void* __restrict__ Cp,
              int N, int K,
              long sAm, long sAk, long batchA,
              long sBk, long sBn, long batchB, long batchC)
{
    __shared__ __align__(16) __bf16 As[2][BT][LDSS];
    __shared__ __align__(16) __bf16 Bs[2][BT][LDSS];   // stored [n][k]

    const int t    = threadIdx.x;
    const int lane = t & 31;
    const int wave = t >> 5;
    const int wm   = (wave >> 2) * 64;   // wave row offset in tile
    const int wn   = (wave & 3) * 32;    // wave col offset in tile
    const int l15  = lane & 15;
    const int h    = lane >> 4;

    const long m0 = (long)blockIdx.y * BT;
    const long n0 = (long)blockIdx.x * BT;
    A  += (long)blockIdx.z * batchA;
    Bm += (long)blockIdx.z * batchB;

    v8f acc[4][2] = {};

    // prologue: stage first K-tile into buffer 0
    stage_tile<AKC>(A,  sAm, sAk, As[0], m0, 0, t);
    stage_tile<BKC>(Bm, sBn, sBk, Bs[0], n0, 0, t);

    const int nk = K / KT;
    for (int kt = 0; kt < nk; ++kt) {
        if (AKC || BKC) wait_async0();   // own async copies done
        __syncthreads();                 // everyone's copies visible

        // overlap: stage next tile into the other buffer while computing
        if (kt + 1 < nk) {
            stage_tile<AKC>(A,  sAm, sAk, As[(kt + 1) & 1], m0, (long)(kt + 1) * KT, t);
            stage_tile<BKC>(Bm, sBn, sBk, Bs[(kt + 1) & 1], n0, (long)(kt + 1) * KT, t);
        }

        const __bf16 (*Asb)[LDSS] = As[kt & 1];
        const __bf16 (*Bsb)[LDSS] = Bs[kt & 1];

        #pragma unroll
        for (int ks = 0; ks < KT; ks += 32) {
            // B 32x16 frag: lane(l15,h): col=l15, k in [8h,8h+8) U [16+8h,+8)
            v16bf bfrag[2];
            #pragma unroll
            for (int tn = 0; tn < 2; ++tn) {
                const __bf16* bp = &Bsb[wn + tn * 16 + l15][ks];
                const v8bf lo = *(const v8bf*)(bp + 8 * h);
                const v8bf hi = *(const v8bf*)(bp + 16 + 8 * h);
                #pragma unroll
                for (int j = 0; j < 8; ++j) { bfrag[tn][j] = lo[j]; bfrag[tn][8 + j] = hi[j]; }
            }
            // A 16x32 frag: lane(l15,h): row=l15, k in [8h,8h+8) U [16+8h,+8)
            #pragma unroll
            for (int tm = 0; tm < 4; ++tm) {
                const __bf16* ap = &Asb[wm + tm * 16 + l15][ks];
                const v8bf lo = *(const v8bf*)(ap + 8 * h);
                const v8bf hi = *(const v8bf*)(ap + 16 + 8 * h);
                v16bf af;
                #pragma unroll
                for (int j = 0; j < 8; ++j) { af[j] = lo[j]; af[8 + j] = hi[j]; }
                acc[tm][0] = __builtin_amdgcn_wmma_f32_16x16x32_bf16(
                    false, af, false, bfrag[0], (short)0, acc[tm][0], false, false);
                acc[tm][1] = __builtin_amdgcn_wmma_f32_16x16x32_bf16(
                    false, af, false, bfrag[1], (short)0, acc[tm][1], false, false);
            }
        }
    }

    // epilogue: bias + store (C/D layout: VGPR r -> row r+8h, col = l15)
    const long zC = (long)blockIdx.z * batchC;
    #pragma unroll
    for (int tn = 0; tn < 2; ++tn) {
        const long col = n0 + wn + tn * 16 + l15;
        const float bv = bias ? bias[col] : 0.0f;
        #pragma unroll
        for (int tm = 0; tm < 4; ++tm) {
            #pragma unroll
            for (int r = 0; r < 8; ++r) {
                const long row = m0 + wm + tm * 16 + r + 8 * h;
                const float val = acc[tm][tn][r] + bv;
                if (OUTBF) ((__bf16*)Cp)[zC + row * (long)N + col] = (__bf16)val;
                else       ((float*)Cp)[zC + row * (long)N + col] = val;
            }
        }
    }
}

// ---------------------------------------------------------------------------
// f32 -> bf16 elementwise convert (vectorized). n must be a multiple of 1024.
// ---------------------------------------------------------------------------
__global__ __launch_bounds__(256)
void ita_cvt_bf16(const float* __restrict__ x, __bf16* __restrict__ y, long n)
{
    const long i = ((long)blockIdx.x * 256 + threadIdx.x) * 4;
    if (i >= n) return;
    const float4 v = *(const float4*)(x + i);
    v4bf o; o[0] = (__bf16)v.x; o[1] = (__bf16)v.y; o[2] = (__bf16)v.z; o[3] = (__bf16)v.w;
    *(v4bf*)(y + i) = o;
}

// ---------------------------------------------------------------------------
// Softmax over last dim: f32 logits in, bf16 normalized out. 1 block / row.
// ---------------------------------------------------------------------------
__global__ __launch_bounds__(256)
void ita_softmax_rows(const float* __restrict__ X, __bf16* __restrict__ Y, int ncols)
{
    __shared__ float red[256];
    const long row = blockIdx.x;
    const float* p = X + row * (long)ncols;
    __bf16*      q = Y + row * (long)ncols;
    const int t = threadIdx.x;

    float m = -INFINITY;
    for (int c = t; c < ncols; c += 256) m = fmaxf(m, p[c]);
    red[t] = m; __syncthreads();
    for (int s = 128; s > 0; s >>= 1) { if (t < s) red[t] = fmaxf(red[t], red[t + s]); __syncthreads(); }
    m = red[0]; __syncthreads();

    float sum = 0.0f;
    for (int c = t; c < ncols; c += 256) sum += __expf(p[c] - m);
    red[t] = sum; __syncthreads();
    for (int s = 128; s > 0; s >>= 1) { if (t < s) red[t] += red[t + s]; __syncthreads(); }
    const float inv = 1.0f / red[0];

    for (int c = t; c < ncols; c += 256) q[c] = (__bf16)(__expf(p[c] - m) * inv);
}

// ---------------------------------------------------------------------------
// Softmax over K (row) dim of X[b,K,D]: f32 in, bf16 out.
// Block: 256 = 64 cols x 4 k-slices. Grid: (D/64, B).
// ---------------------------------------------------------------------------
__global__ __launch_bounds__(256)
void ita_softmax_cols(const float* __restrict__ X, __bf16* __restrict__ Y,
                      int Kdim, int Ddim)
{
    __shared__ float red[4][64];
    const int t  = threadIdx.x;
    const int cx = t & 63;
    const int kg = t >> 6;
    const long d    = (long)blockIdx.x * 64 + cx;
    const long boff = (long)blockIdx.y * (long)Kdim * Ddim + d;
    const float* bx = X + boff;
    __bf16*      by = Y + boff;

    float m = -INFINITY;
    for (int k = kg; k < Kdim; k += 4) m = fmaxf(m, bx[(long)k * Ddim]);
    red[kg][cx] = m; __syncthreads();
    if (kg == 0)
        red[0][cx] = fmaxf(fmaxf(red[0][cx], red[1][cx]), fmaxf(red[2][cx], red[3][cx]));
    __syncthreads();
    m = red[0][cx]; __syncthreads();

    float s = 0.0f;
    for (int k = kg; k < Kdim; k += 4) s += __expf(bx[(long)k * Ddim] - m);
    red[kg][cx] = s; __syncthreads();
    if (kg == 0)
        red[0][cx] = red[0][cx] + red[1][cx] + red[2][cx] + red[3][cx];
    __syncthreads();
    const float inv = 1.0f / red[0][cx];

    for (int k = kg; k < Kdim; k += 4)
        by[(long)k * Ddim] = (__bf16)(__expf(bx[(long)k * Ddim] - m) * inv);
}

// ---------------------------------------------------------------------------
extern "C" void kernel_launch(void* const* d_in, const int* in_sizes, int n_in,
                              void* d_out, int out_size, void* d_ws, size_t ws_size,
                              hipStream_t stream)
{
    const float* latent = (const float*)d_in[0];   // [4, 2048, 1024]
    const float* input  = (const float*)d_in[1];   // [4, 2048, 1024]
    const float* Wl = (const float*)d_in[2];  const float* bl = (const float*)d_in[3];
    const float* Wu = (const float*)d_in[4];  const float* bu = (const float*)d_in[5];
    const float* WA = (const float*)d_in[6];  const float* bA = (const float*)d_in[7];
    const float* WV = (const float*)d_in[8];  const float* bV = (const float*)d_in[9];
    const float* Wo = (const float*)d_in[10]; const float* bo = (const float*)d_in[11];
    float* out = (float*)d_out;

    const long D = 1024, Bn = 4, TQ = 2048, TKV = 2048;
    const long MT  = Bn * TQ;     // 8192 flattened rows
    const long SEQ = TQ * D;      // per-batch stride of [T, D]

    // ---- workspace carve-up ----
    char* w = (char*)d_ws;
    float*  logits = (float*)w;        w += MT * D * sizeof(float);   // 32 MB (shared by both softmax stages)
    __bf16* lat_b  = (__bf16*)w;       w += MT * D * 2;
    __bf16* inp_b  = (__bf16*)w;       w += MT * D * 2;
    __bf16* wl_b   = (__bf16*)w;       w += D * D * 2;
    __bf16* wu_b   = (__bf16*)w;       w += D * D * 2;
    __bf16* wa_b   = (__bf16*)w;       w += D * D * 2;
    __bf16* wv_b   = (__bf16*)w;       w += D * D * 2;
    __bf16* wo_b   = (__bf16*)w;       w += D * D * 2;
    __bf16* t0     = (__bf16*)w;       w += MT * D * 2;   // emb0_bf, later Dm_bf
    __bf16* t1     = (__bf16*)w;       w += MT * D * 2;   // l_bf, later A_bf
    __bf16* t2     = (__bf16*)w;       w += MT * D * 2;   // ov_bf
    __bf16* Mb     = (__bf16*)w;       w += Bn * D * D * 2; // l @ ov, [B, d, d]

    const dim3 blk(256);

    // ---- 0. convert operands to bf16 once ----
    auto cvt = [&](const float* x, __bf16* y, long n) {
        ita_cvt_bf16<<<dim3((unsigned)(n / 1024)), blk, 0, stream>>>(x, y, n);
    };
    cvt(latent, lat_b, MT * D);
    cvt(input,  inp_b, MT * D);
    cvt(Wl, wl_b, D * D);  cvt(Wu, wu_b, D * D);  cvt(WA, wa_b, D * D);
    cvt(WV, wv_b, D * D);  cvt(Wo, wo_b, D * D);

    const dim3 gW((unsigned)(D / BT), (unsigned)(MT / BT), 1);  // 8192x1024x1024 weight GEMMs

    // 1. emb0 = latent @ Wl^T + bl                 -> t0 (bf16)
    ita_gemm<true, true, true><<<gW, blk, 0, stream>>>(
        lat_b, wl_b, bl, t0, (int)D, (int)D,  D, 1, 0,  1, D, 0,  0);

    // 2. l_logits = emb0 @ Wu^T + bu -> logits ; row softmax over d -> t1 (bf16)
    ita_gemm<true, true, false><<<gW, blk, 0, stream>>>(
        t0, wu_b, bu, logits, (int)D, (int)D,  D, 1, 0,  1, D, 0,  0);
    ita_softmax_rows<<<dim3((unsigned)MT), blk, 0, stream>>>(logits, t1, (int)D);

    // 3. ov = latent @ Wo^T + bo                   -> t2 (bf16)
    ita_gemm<true, true, true><<<gW, blk, 0, stream>>>(
        lat_b, wo_b, bo, t2, (int)D, (int)D,  D, 1, 0,  1, D, 0,  0);

    // 4. M[b] = l^T @ ov  (d x d)  [fusion: o = A @ (l @ ov)]   -> Mb (bf16)
    //    A(e,q) = t1[b, q, e]  (sAm=1, sAk=D) ; B(q,d') = t2[b, q, d'] (sBk=D, sBn=1)
    ita_gemm<false, false, true><<<dim3((unsigned)(D / BT), (unsigned)(D / BT), (unsigned)Bn), blk, 0, stream>>>(
        t1, t2, nullptr, Mb, (int)D, (int)TQ,  1, D, SEQ,  D, 1, SEQ,  D * D);

    // 5. Dm = input @ WA^T + bA                    -> t0 (bf16)
    ita_gemm<true, true, true><<<gW, blk, 0, stream>>>(
        inp_b, wa_b, bA, t0, (int)D, (int)D,  D, 1, 0,  1, D, 0,  0);

    // 6. A_logits = Dm @ WV^T + bV -> logits ; softmax over T_kv -> t1 (bf16)
    ita_gemm<true, true, false><<<gW, blk, 0, stream>>>(
        t0, wv_b, bV, logits, (int)D, (int)D,  D, 1, 0,  1, D, 0,  0);
    ita_softmax_cols<<<dim3((unsigned)(D / 64), (unsigned)Bn), blk, 0, stream>>>(
        logits, t1, (int)TKV, (int)D);

    // 7. o[b] = A[b] @ M[b]                        -> d_out (f32)
    //    A(k,e) = t1[b, k, e] (sAm=D, sAk=1) ; B(e,d') = Mb[b, e, d'] (sBk=D, sBn=1)
    ita_gemm<true, false, false><<<dim3((unsigned)(D / BT), (unsigned)(TKV / BT), (unsigned)Bn), blk, 0, stream>>>(
        t1, Mb, nullptr, out, (int)D, (int)D,  D, 1, SEQ,  D, 1, D * D,  (long)TKV * D);
}
